// GATNet_40072044871759
// MI455X (gfx1250) — matrix operand order repaired and checked by reading, not compiled, not run on hardware
//
#include <hip/hip_runtime.h>
#include <hip/hip_bf16.h>
#include <math.h>

// ---------------- types for WMMA ----------------
typedef __attribute__((ext_vector_type(16))) __bf16 v16bf;
typedef __attribute__((ext_vector_type(8)))  float  v8f;

#define D_DIM 128

// ---------------- init: zero out (float4), segmax_enc=0, denom=0 ----------------
__global__ void gat_init_kernel(float4* __restrict__ out4, int outn4,
                                unsigned* __restrict__ senc, float* __restrict__ denom, int n) {
    int i = blockIdx.x * blockDim.x + threadIdx.x;
    int stride = gridDim.x * blockDim.x;
    float4 z = make_float4(0.f, 0.f, 0.f, 0.f);
    for (int j = i; j < outn4; j += stride) out4[j] = z;
    for (int j = i; j < n; j += stride) { senc[j] = 0u; denom[j] = 0.0f; }
}

// ---------------- GEMM: h = x @ W via v_wmma_f32_16x16x32_bf16 ----------------
// Block = 256 threads = 8 waves. Each wave computes rows [r0, r0+16) x all 128 cols.
// W staged transposed in LDS (bf16) once per block.
__global__ __launch_bounds__(256) void gat_gemm_kernel(const float* __restrict__ x,
                                                       const float* __restrict__ W,
                                                       float* __restrict__ h, int nnodes) {
    __shared__ __bf16 Wt[D_DIM][D_DIM + 8];  // transposed, padded to avoid bank conflicts

    int tid = threadIdx.x;
    for (int i = tid; i < D_DIM * D_DIM; i += 256) {
        int k = i >> 7;
        int c = i & 127;
        Wt[c][k] = (__bf16)W[i];
    }
    __syncthreads();

    int wave = tid >> 5;
    int lane = tid & 31;
    int r0 = blockIdx.x * 128 + wave * 16;
    if (r0 >= nnodes) return;

    int m   = lane & 15;               // row within tile (A) / col within tile (B,C)
    int kba = (lane < 16) ? 0 : 8;     // A K-base: chunks [kba,kba+8) and [kba+16,kba+24)
    int kbb = (lane < 16) ? 0 : 8;     // B K-base: chunks [kbb,kbb+8) and [kbb+16,kbb+24)

    int row  = r0 + m;
    int rowc = (row < nnodes) ? row : (nnodes - 1);  // clamp for safe loads

    v8f acc[8];
    #pragma unroll
    for (int n = 0; n < 8; ++n) acc[n] = (v8f){0.f,0.f,0.f,0.f,0.f,0.f,0.f,0.f};

    for (int k0 = 0; k0 < D_DIM; k0 += 32) {
        // ---- A fragment: rows of x, converted f32 -> bf16 ----
        v16bf a;
        const float* xp = x + (size_t)rowc * D_DIM + k0 + kba;
        #pragma unroll
        for (int i = 0; i < 8; ++i)  a[i] = (__bf16)xp[i];        // K = k0+kba+i
        #pragma unroll
        for (int i = 8; i < 16; ++i) a[i] = (__bf16)xp[i + 8];    // K = k0+kba+16+(i-8)

        #pragma unroll
        for (int n = 0; n < 8; ++n) {
            // ---- B fragment: column (n*16+m) of W, K = k0+kbb+i (i<8), k0+kbb+16+(i-8) (i>=8) ----
            v16bf b;
            const __bf16* wp = &Wt[n * 16 + m][k0 + kbb];
            #pragma unroll
            for (int i = 0; i < 16; ++i) b[i] = wp[(i < 8) ? i : (i + 8)];
            acc[n] = __builtin_amdgcn_wmma_f32_16x16x32_bf16(
                false, a, false, b, (short)0, acc[n], false, false);
        }
    }

    // ---- write D per documented layout: VGPR r holds M=r (lanes<16) / M=8+r (lanes>=16) ----
    int mbase = (lane < 16) ? 0 : 8;
    float* hout = h + (size_t)(r0 + mbase) * D_DIM + m;   // row (r0+mbase), col (n*16+m)
    if (r0 + 16 <= nnodes) {
        // fast path: whole 16-row tile in range -> 64 unguarded stores, no exec churn
        #pragma unroll
        for (int r = 0; r < 8; ++r) {
            #pragma unroll
            for (int n = 0; n < 8; ++n) {
                hout[(size_t)r * D_DIM + n * 16] = acc[n][r];
            }
        }
    } else {
        #pragma unroll
        for (int r = 0; r < 8; ++r) {
            if (r0 + mbase + r < nnodes) {
                #pragma unroll
                for (int n = 0; n < 8; ++n) {
                    hout[(size_t)r * D_DIM + n * 16] = acc[n][r];
                }
            }
        }
    }
}

// ---------------- scores: si = h . a_i, sj = h . a_j (one wave per node) ----------------
__global__ __launch_bounds__(256) void gat_scores_kernel(const float* __restrict__ h,
                                                         const float* __restrict__ att,
                                                         float* __restrict__ si,
                                                         float* __restrict__ sj, int n) {
    int node = blockIdx.x * (blockDim.x >> 5) + (threadIdx.x >> 5);
    if (node >= n) return;
    int lane = threadIdx.x & 31;
    const float* hp = h + (size_t)node * D_DIM;
    float pi = 0.f, pj = 0.f;
    #pragma unroll
    for (int c = lane; c < D_DIM; c += 32) {
        float v = hp[c];
        pi += v * att[c];
        pj += v * att[D_DIM + c];
    }
    #pragma unroll
    for (int o = 16; o > 0; o >>= 1) {
        pi += __shfl_down(pi, o, 32);
        pj += __shfl_down(pj, o, 32);
    }
    if (lane == 0) { si[node] = pi; sj[node] = pj; }
}

// float -> monotone unsigned encoding (for integer atomicMax on floats incl. negatives)
__device__ __forceinline__ unsigned f32_enc(float f) {
    unsigned b = __float_as_uint(f);
    return (b & 0x80000000u) ? ~b : (b | 0x80000000u);
}
__device__ __forceinline__ float f32_dec(unsigned e) {
    unsigned b = (e & 0x80000000u) ? (e & 0x7fffffffu) : ~e;
    return __uint_as_float(b);
}

// ---------------- per edge: alpha = leaky_relu(si[dst]+sj[src])*w; segment max over dst ----------------
__global__ void gat_edge_alpha_kernel(const int* __restrict__ ei, const float* __restrict__ ew,
                                      const float* __restrict__ si, const float* __restrict__ sj,
                                      float* __restrict__ alpha, unsigned* __restrict__ senc, int ne) {
    int e = blockIdx.x * blockDim.x + threadIdx.x;
    if (e >= ne) return;
    int s = ei[e];
    int d = ei[ne + e];
    float v = si[d] + sj[s];
    v = (v > 0.f) ? v : 0.2f * v;   // leaky_relu slope 0.2
    v *= ew[e];
    alpha[e] = v;
    atomicMax(&senc[d], f32_enc(v));
}

// ---------------- per edge: aexp = exp(alpha - segmax[dst]); denom[dst] += aexp ----------------
__global__ void gat_edge_exp_kernel(const int* __restrict__ ei, const float* __restrict__ alpha,
                                    const unsigned* __restrict__ senc, float* __restrict__ aexp,
                                    float* __restrict__ denom, int ne) {
    int e = blockIdx.x * blockDim.x + threadIdx.x;
    if (e >= ne) return;
    int d = ei[ne + e];
    float m = f32_dec(senc[d]);
    float a = __expf(alpha[e] - m);
    aexp[e] = a;
    atomicAdd(&denom[d], a);
}

// ---------------- aggregation: out[dst] += (aexp/denom[dst]) * h[src]; one wave per edge ----------------
__global__ __launch_bounds__(256) void gat_edge_aggr_kernel(const int* __restrict__ ei,
                                                            const float* __restrict__ aexp,
                                                            const float* __restrict__ denom,
                                                            const float* __restrict__ h,
                                                            float* __restrict__ out, int ne) {
    int e = blockIdx.x * (blockDim.x >> 5) + (threadIdx.x >> 5);
    if (e >= ne) return;
    int lane = threadIdx.x & 31;
    int s = ei[e];
    int d = ei[ne + e];
    float coef = aexp[e] / (denom[d] + 1e-16f);
    const float4* hp = (const float4*)(h + (size_t)s * D_DIM);
    float4 v = hp[lane];                        // coalesced 512B per wave, L2-resident
    float* op = out + (size_t)d * D_DIM + lane * 4;
    atomicAdd(op + 0, v.x * coef);
    atomicAdd(op + 1, v.y * coef);
    atomicAdd(op + 2, v.z * coef);
    atomicAdd(op + 3, v.w * coef);
}

// ---------------- final relu ----------------
__global__ void gat_relu_kernel(float* __restrict__ out, int n) {
    int i = blockIdx.x * blockDim.x + threadIdx.x;
    int stride = gridDim.x * blockDim.x;
    for (int j = i; j < n; j += stride) {
        float v = out[j];
        out[j] = (v > 0.f) ? v : 0.f;
    }
}

extern "C" void kernel_launch(void* const* d_in, const int* in_sizes, int n_in,
                              void* d_out, int out_size, void* d_ws, size_t ws_size,
                              hipStream_t stream) {
    const float* x   = (const float*)d_in[0];   // [N,128]
    const int*   ei  = (const int*)d_in[1];     // [2,NE]
    const float* ew  = (const float*)d_in[2];   // [NE]
    const float* W   = (const float*)d_in[3];   // [128,128]
    const float* att = (const float*)d_in[4];   // [256]

    const int N  = in_sizes[0] / D_DIM;
    const int NE = in_sizes[1] / 2;

    float* out = (float*)d_out;                 // [N,128]

    // workspace layout (floats)
    float* ws = (float*)d_ws;
    float*    h     = ws;                              // N*128
    float*    si    = h + (size_t)N * D_DIM;           // N
    float*    sj    = si + N;                          // N
    float*    alpha = sj + N;                          // NE
    float*    aexp  = alpha + NE;                      // NE
    unsigned* senc  = (unsigned*)(aexp + NE);          // N
    float*    denom = (float*)(senc + N);              // N

    // 1. init (out=0, senc=0, denom=0)
    gat_init_kernel<<<4096, 256, 0, stream>>>((float4*)out, N * (D_DIM / 4), senc, denom, N);

    // 2. h = x @ W (WMMA bf16, fp32 accumulate)
    int gemm_blocks = (N + 127) / 128;
    gat_gemm_kernel<<<gemm_blocks, 256, 0, stream>>>(x, W, h, N);

    // 3. attention scores
    int sc_blocks = (N + 7) / 8;
    gat_scores_kernel<<<sc_blocks, 256, 0, stream>>>(h, att, si, sj, N);

    // 4. edge alpha + segment max
    int eb = (NE + 255) / 256;
    gat_edge_alpha_kernel<<<eb, 256, 0, stream>>>(ei, ew, si, sj, alpha, senc, NE);

    // 5. exp + segment sum (denominator)
    gat_edge_exp_kernel<<<eb, 256, 0, stream>>>(ei, alpha, senc, aexp, denom, NE);

    // 6. aggregation (wave per edge)
    int ab = (NE + 7) / 8;
    gat_edge_aggr_kernel<<<ab, 256, 0, stream>>>(ei, aexp, denom, h, out, NE);

    // 7. relu
    gat_relu_kernel<<<4096, 256, 0, stream>>>(out, N * D_DIM);
}
